// AsymQuantMatMul_89232240542109
// MI455X (gfx1250) — 8-bit matrix operand order repaired and checked by reading, compiled, not run on hardware
//
#include <hip/hip_runtime.h>

typedef __attribute__((ext_vector_type(8))) int v8i;

#define BS 4
#define H  1024
#define W  4096
#define QMAXF 15.0f

__global__ void kminmax(const float* __restrict__ A, const float* __restrict__ B,
                        float* __restrict__ pmn, float* __restrict__ pmx) {
    const float* X = (blockIdx.z == 0) ? A : B;
    const float4* base = (const float4*)(X + (size_t)blockIdx.y * H * W) +
                         (size_t)blockIdx.x * 16384;
    float mn =  3.4e38f, mx = -3.4e38f;
    for (int i = threadIdx.x; i < 16384; i += 256) {
        float4 v = base[i];
        mn = fminf(mn, fminf(fminf(v.x, v.y), fminf(v.z, v.w)));
        mx = fmaxf(mx, fmaxf(fmaxf(v.x, v.y), fmaxf(v.z, v.w)));
    }
    __shared__ float smn[256], smx[256];
    int t = threadIdx.x;
    smn[t] = mn; smx[t] = mx;
    __syncthreads();
    for (int s = 128; s > 0; s >>= 1) {
        if (t < s) { smn[t] = fminf(smn[t], smn[t + s]); smx[t] = fmaxf(smx[t], smx[t + s]); }
        __syncthreads();
    }
    if (t == 0) {
        int idx = (blockIdx.z * BS + blockIdx.y) * 64 + blockIdx.x;
        pmn[idx] = smn[0];
        pmx[idx] = smx[0];
    }
}

__global__ void kfinalize(const float* __restrict__ pmn, const float* __restrict__ pmx,
                          float* __restrict__ sz) {
    int g = blockIdx.x;
    int t = threadIdx.x;
    __shared__ float smn[64], smx[64];
    smn[t] = pmn[g * 64 + t];
    smx[t] = pmx[g * 64 + t];
    __syncthreads();
    for (int s = 32; s > 0; s >>= 1) {
        if (t < s) { smn[t] = fminf(smn[t], smn[t + s]); smx[t] = fmaxf(smx[t], smx[t + s]); }
        __syncthreads();
    }
    if (t == 0) {
        float mn = smn[0], mx = smx[0];
        float scale = (mx - mn) / QMAXF;
        if (scale == 0.0f) scale = 1.0f;
        float zero = fminf(fmaxf(rintf(-mn / scale), 0.0f), QMAXF);
        sz[g * 2 + 0] = scale;
        sz[g * 2 + 1] = zero;
    }
}

__global__ void kquant(const float* __restrict__ A, const float* __restrict__ B,
                       unsigned char* __restrict__ qA, unsigned char* __restrict__ qB,
                       float* __restrict__ rA, float* __restrict__ rB,
                       const float* __restrict__ sz) {
    int tensor = blockIdx.z, b = blockIdx.y, row = blockIdx.x, t = threadIdx.x;
    const float* X = (tensor ? B : A) + ((size_t)b * H + row) * W;
    unsigned int* Q = (unsigned int*)((tensor ? qB : qA) + ((size_t)b * H + row) * W);
    float* R = tensor ? rB : rA;
    float scale = sz[(tensor * BS + b) * 2 + 0];
    float zero  = sz[(tensor * BS + b) * 2 + 1];
    float inv = 1.0f / scale;
    float lsum = 0.0f;
    const float4* x4 = (const float4*)X;
    for (int i = t; i < W / 4; i += 256) {
        float4 v = x4[i];
        float q0 = fminf(fmaxf(rintf(v.x * inv) + zero, 0.0f), QMAXF);
        float q1 = fminf(fmaxf(rintf(v.y * inv) + zero, 0.0f), QMAXF);
        float q2 = fminf(fmaxf(rintf(v.z * inv) + zero, 0.0f), QMAXF);
        float q3 = fminf(fmaxf(rintf(v.w * inv) + zero, 0.0f), QMAXF);
        lsum += q0 + q1 + q2 + q3;
        unsigned int p = (unsigned int)(int)q0 |
                         ((unsigned int)(int)q1 << 8) |
                         ((unsigned int)(int)q2 << 16) |
                         ((unsigned int)(int)q3 << 24);
        Q[i] = p;
    }
    __shared__ float ss[256];
    ss[t] = lsum;
    __syncthreads();
    for (int s = 128; s > 0; s >>= 1) {
        if (t < s) ss[t] += ss[t + s];
        __syncthreads();
    }
    if (t == 0) R[(size_t)b * H + row] = ss[0];
}

__global__ void __launch_bounds__(256)
kgemm(const unsigned char* __restrict__ qA, const unsigned char* __restrict__ qB,
      const float* __restrict__ rA, const float* __restrict__ rB,
      const float* __restrict__ sz, float* __restrict__ out) {
    const int b  = blockIdx.z;
    const int bm = blockIdx.y;
    const int bn = blockIdx.x;
    const int t    = threadIdx.x;
    const int lane = t & 31;
    const int wave = t >> 5;
    const int wm   = wave >> 2;
    const int wn   = wave & 3;

    __shared__ unsigned char ldsA[128 * 64];
    __shared__ unsigned char ldsB[128 * 64];

    const unsigned char* gA = qA + ((size_t)b * H + (size_t)bm * 128) * W;
    const unsigned char* gB = qB + ((size_t)b * H + (size_t)bn * 128) * W;

    v8i acc[4][2];
#pragma unroll
    for (int i = 0; i < 4; ++i)
#pragma unroll
        for (int j = 0; j < 2; ++j) { v8i z = {}; acc[i][j] = z; }

    const int lrow = t >> 2;
    const int lcol = (t & 3) * 16;
    const int r  = lane & 15;
    const int kh = lane >> 4;

    for (int kb = 0; kb < W / 64; ++kb) {
        __syncthreads();
#pragma unroll
        for (int pass = 0; pass < 2; ++pass) {
            int row = pass * 64 + lrow;
            *(uint4*)(ldsA + row * 64 + lcol) =
                *(const uint4*)(gA + (size_t)row * W + kb * 64 + lcol);
            *(uint4*)(ldsB + row * 64 + lcol) =
                *(const uint4*)(gB + (size_t)row * W + kb * 64 + lcol);
        }
        __syncthreads();

        v8i afrag[4];
#pragma unroll
        for (int i = 0; i < 4; ++i) {
            const uint2* p = (const uint2*)(ldsA + (wm * 64 + i * 16 + r) * 64 + kh * 16);
            uint2 a0 = p[0], a1 = p[1], a2 = p[4], a3 = p[5];
            v8i f;
            f[0] = (int)a0.x; f[1] = (int)a0.y;
            f[2] = (int)a1.x; f[3] = (int)a1.y;
            f[4] = (int)a2.x; f[5] = (int)a2.y;
            f[6] = (int)a3.x; f[7] = (int)a3.y;
            afrag[i] = f;
        }

        v8i bfrag[2];
#pragma unroll
        for (int j = 0; j < 2; ++j) {
            const unsigned char* base = ldsB + (wn * 32 + j * 16 + r) * 64;
            uint4 u0 = *(const uint4*)(base + kh * 16);
            uint4 u1 = *(const uint4*)(base + 32 + kh * 16);
            v8i f;
            f[0] = (int)u0.x; f[1] = (int)u0.y; f[2] = (int)u0.z; f[3] = (int)u0.w;
            f[4] = (int)u1.x; f[5] = (int)u1.y; f[6] = (int)u1.z; f[7] = (int)u1.w;
            bfrag[j] = f;
        }

#pragma unroll
        for (int i = 0; i < 4; ++i)
#pragma unroll
            for (int j = 0; j < 2; ++j)
                acc[i][j] = __builtin_amdgcn_wmma_i32_16x16x64_iu8(
                     false, afrag[i],
                     false, bfrag[j],
                    acc[i][j],  false,  false);
    }

    const float sA = sz[b * 2 + 0],        zA = sz[b * 2 + 1];
    const float sB = sz[(BS + b) * 2 + 0], zB = sz[(BS + b) * 2 + 1];
    const float alpha = sA * sB;
    const float beta  = -alpha * zB;
    const float gamma = -alpha * zA;
    const float delta = alpha * (float)W * zA * zB;

    float* C = out + (size_t)b * H * H;
#pragma unroll
    for (int i = 0; i < 4; ++i) {
#pragma unroll
        for (int j = 0; j < 2; ++j) {
            int gcol = bn * 128 + wn * 32 + j * 16 + (lane & 15);
            float rb = rB[(size_t)b * H + gcol];
#pragma unroll
            for (int v = 0; v < 8; ++v) {
                int grow = bm * 128 + wm * 64 + i * 16 + v + ((lane >> 4) << 3);
                float ra = rA[(size_t)b * H + grow];
                C[(size_t)grow * H + gcol] =
                    alpha * (float)acc[i][j][v] + beta * ra + gamma * rb + delta;
            }
        }
    }
}

extern "C" void kernel_launch(void* const* d_in, const int* in_sizes, int n_in,
                              void* d_out, int out_size, void* d_ws, size_t ws_size,
                              hipStream_t stream) {
    (void)in_sizes; (void)n_in; (void)out_size; (void)ws_size;

    const float* A = (const float*)d_in[0];
    const float* B = (const float*)d_in[1];
    float* out = (float*)d_out;

    unsigned char* qA = (unsigned char*)d_ws;
    unsigned char* qB = qA + (size_t)BS * H * W;
    float* rA  = (float*)(qB + (size_t)BS * H * W);
    float* rB  = rA + (size_t)BS * H;
    float* pmn = rB + (size_t)BS * H;
    float* pmx = pmn + 2 * BS * 64;
    float* sz  = pmx + 2 * BS * 64;

    kminmax  <<<dim3(64, BS, 2), 256, 0, stream>>>(A, B, pmn, pmx);
    kfinalize<<<dim3(2 * BS),     64, 0, stream>>>(pmn, pmx, sz);
    kquant   <<<dim3(H, BS, 2),  256, 0, stream>>>(A, B, qA, qB, rA, rB, sz);
    kgemm    <<<dim3(H / 128, H / 128, BS), 256, 0, stream>>>(qA, qB, rA, rB, sz, out);
}
